// self_attn_8074538516679
// MI455X (gfx1250) — matrix operand order repaired and checked by reading, not compiled, not run on hardware
//
#include <hip/hip_runtime.h>

// ---------------------------------------------------------------------------
// Fused sigmoid-gated self-attention for MI455X (gfx1250, wave32, WMMA).
// B=4, C=128, C8=16, N=64*64=4096.
//
//   q = Wq x + bq  (16xN)   k = Wk x + bk (16xN)   v = Wv x + bv (128xN)
//   out[c,n] = gamma * sum_m sigmoid(q[:,n].k[:,m]) * v[c,m] + x[c,n]
//
// attn (4 * 4096^2 f32 = 268MB) is never materialized: flash-style fusion.
// K/V chunk staging uses CDNA5 GLOBAL_LOAD_ASYNC_TO_LDS (ASYNCcnt) with
// double-buffered LDS when the builtins are available.
// ---------------------------------------------------------------------------

typedef _Float16 half8  __attribute__((ext_vector_type(8)));
typedef _Float16 half16 __attribute__((ext_vector_type(16)));
typedef float    f32x8  __attribute__((ext_vector_type(8)));
typedef int      v2i    __attribute__((ext_vector_type(2)));
typedef int      v4i    __attribute__((ext_vector_type(4)));

#define NPOS 4096
#define CCH  128
#define CQ   16

#define GLOBAL_AS __attribute__((address_space(1)))
#define LDS_AS    __attribute__((address_space(3)))

#if defined(__has_builtin)
#if __has_builtin(__builtin_amdgcn_global_load_async_to_lds_b128) && \
    __has_builtin(__builtin_amdgcn_global_load_async_to_lds_b64) &&  \
    __has_builtin(__builtin_amdgcn_s_wait_asynccnt)
#define USE_ASYNC_LDS 1
#endif
#endif
#ifndef USE_ASYNC_LDS
#define USE_ASYNC_LDS 0
#endif

__device__ __forceinline__ half16 cat16(half8 lo, half8 hi) {
  half16 r;
#pragma unroll
  for (int i = 0; i < 8; ++i) { r[i] = lo[i]; r[i + 8] = hi[i]; }
  return r;
}

// ---------------------------------------------------------------------------
// Kernel 1: projections q/k/v via WMMA, f16 outputs in WMMA-friendly layouts.
// Grid: B * (N/64) blocks of 256 threads (8 waves). Per block: 64 positions.
// ---------------------------------------------------------------------------
__launch_bounds__(256)
__global__ void proj_kernel(const float* __restrict__ x,
                            const float* __restrict__ Wq, const float* __restrict__ bq,
                            const float* __restrict__ Wk, const float* __restrict__ bk,
                            const float* __restrict__ Wv, const float* __restrict__ bv,
                            _Float16* __restrict__ qh,   // [B][N][16]  (n-major)
                            _Float16* __restrict__ kh,   // [B][N][16]  (m-major)
                            _Float16* __restrict__ vh)   // [B][C][N]
{
  const int b    = blockIdx.x >> 6;
  const int n0   = (blockIdx.x & 63) * 64;
  const int tid  = threadIdx.x;
  const int lane = tid & 31;
  const int w    = tid >> 5;
  const int hl   = lane >> 4;   // half-lane select
  const int l16  = lane & 15;

  constexpr int XS = 136;   // x-tile row stride (halves): 16B aligned, bank-safe
  constexpr int WS = 136;   // weight row stride
  __shared__ __attribute__((aligned(16))) _Float16 xT[64 * XS];     // x^T tile
  __shared__ __attribute__((aligned(16))) _Float16 wqs[16 * WS];
  __shared__ __attribute__((aligned(16))) _Float16 wks[16 * WS];
  __shared__ __attribute__((aligned(16))) _Float16 wvs[128 * WS];
  __shared__ float bqs[16], bks[16], bvs[128];

  // stage x[b, :, n0:n0+64] transposed to f16 (coalesced along n)
  for (int idx = tid; idx < 128 * 64; idx += 256) {
    const int c = idx >> 6, nl = idx & 63;
    xT[nl * XS + c] = (_Float16)x[((size_t)b * CCH + c) * NPOS + n0 + nl];
  }
  for (int idx = tid; idx < 16 * 128; idx += 256) {
    const int o = idx >> 7, c = idx & 127;
    wqs[o * WS + c] = (_Float16)Wq[idx];
    wks[o * WS + c] = (_Float16)Wk[idx];
  }
  for (int idx = tid; idx < 128 * 128; idx += 256) {
    const int o = idx >> 7, c = idx & 127;
    wvs[o * WS + c] = (_Float16)Wv[idx];
  }
  if (tid < 16) { bqs[tid] = bq[tid]; bks[tid] = bk[tid]; }
  if (tid < 128) bvs[tid] = bv[tid];
  __syncthreads();

  // 40 output tiles of 16x16: q(4) + k(4) + v(8x4); 5 per wave.
  for (int t = w; t < 40; t += 8) {
    int wsel, otile, ntile;
    if (t < 4)      { wsel = 0; otile = 0;            ntile = t; }
    else if (t < 8) { wsel = 1; otile = 0;            ntile = t - 4; }
    else            { wsel = 2; otile = (t - 8) >> 2; ntile = (t - 8) & 3; }
    const _Float16* W   = wsel == 0 ? wqs : wsel == 1 ? wks : wvs;
    const float*   bias = wsel == 0 ? bqs : wsel == 1 ? bks : bvs;
    const int o0   = otile * 16;
    const int arow = o0 + l16;           // A-matrix row (output channel)
    const int ncol = ntile * 16 + l16;   // B-matrix column (position)

    f32x8 acc = {};
#pragma unroll
    for (int cb = 0; cb < 128; cb += 32) {
      // A (16o x 32c): lane holds row arow, K = {8*hl..+7, 16+8*hl..+7}
      half8 alo = *(const half8*)&W[arow * WS + cb + 8 * hl];
      half8 ahi = *(const half8*)&W[arow * WS + cb + 16 + 8 * hl];
      // B (32c x 16n): lane holds col ncol, K = {8*hl..+7, 16+8*hl..+7}
      half8 blo = *(const half8*)&xT[ncol * XS + cb + 8 * hl];
      half8 bhi = *(const half8*)&xT[ncol * XS + cb + 16 + 8 * hl];
      acc = __builtin_amdgcn_wmma_f32_16x16x32_f16(false, cat16(alo, ahi),
                                                   false, cat16(blo, bhi),
                                                   (short)0, acc, false, false);
    }
#pragma unroll
    for (int r = 0; r < 8; ++r) acc[r] += bias[o0 + r + 8 * hl];

    if (wsel < 2) {
      _Float16* dst = (wsel == 0) ? qh : kh;
      const int n = n0 + ncol;
#pragma unroll
      for (int r = 0; r < 8; ++r)
        dst[((size_t)b * NPOS + n) * CQ + r + 8 * hl] = (_Float16)acc[r];
    } else {
      const int m = n0 + ncol;
#pragma unroll
      for (int r = 0; r < 8; ++r)
        vh[((size_t)b * CCH + o0 + r + 8 * hl) * NPOS + m] = (_Float16)acc[r];
    }
  }
}

// ---------------------------------------------------------------------------
// Kernel 2: fused sigmoid attention + gamma*out + x.
// Grid: B * (N/128) blocks of 256 threads. Each wave: one 16-row n-tile,
// 16x128 f32 accumulator. m-loop in chunks of 64, double-buffered LDS.
// ---------------------------------------------------------------------------
#define KS 24   // K row stride (halves), 16B aligned, bank-conflict-free
#define VS 72   // V row stride
#define PS 40   // P row stride

// Stage one 64-m chunk of K (64x16 halves) and V (128x64 halves) into LDS.
__device__ __forceinline__ void stage_chunk(const _Float16* __restrict__ kh,
                                            const _Float16* __restrict__ vh,
                                            int b, int mc,
                                            _Float16* Kbuf, _Float16* Vbuf,
                                            int tid) {
  const _Float16* kg = kh + ((size_t)b * NPOS + mc) * CQ + tid * 4;
  const int km = tid >> 2, kq = (tid & 3) * 4;
#if USE_ASYNC_LDS
  __builtin_amdgcn_global_load_async_to_lds_b64(
      (GLOBAL_AS v2i*)kg, (LDS_AS v2i*)&Kbuf[km * KS + kq], 0, 0);
#pragma unroll
  for (int kk = 0; kk < 4; ++kk) {
    const int idx = tid + 256 * kk, c = idx >> 3, u = idx & 7;
    __builtin_amdgcn_global_load_async_to_lds_b128(
        (GLOBAL_AS v4i*)(vh + ((size_t)b * CCH + c) * NPOS + mc + u * 8),
        (LDS_AS v4i*)&Vbuf[c * VS + u * 8], 0, 0);
  }
#else
  *(uint2*)&Kbuf[km * KS + kq] = *(const uint2*)kg;
#pragma unroll
  for (int kk = 0; kk < 4; ++kk) {
    const int idx = tid + 256 * kk, c = idx >> 3, u = idx & 7;
    *(uint4*)&Vbuf[c * VS + u * 8] =
        *(const uint4*)(vh + ((size_t)b * CCH + c) * NPOS + mc + u * 8);
  }
#endif
}

__device__ __forceinline__ void wait_async() {
#if USE_ASYNC_LDS
  __builtin_amdgcn_s_wait_asynccnt(0);
#endif
}

__launch_bounds__(256)
__global__ void attn_kernel(const float* __restrict__ x,
                            const _Float16* __restrict__ qh,
                            const _Float16* __restrict__ kh,
                            const _Float16* __restrict__ vh,
                            const float* __restrict__ gamma,
                            float* __restrict__ out)
{
  const int b    = blockIdx.x >> 5;
  const int n0   = (blockIdx.x & 31) * 128;
  const int tid  = threadIdx.x;
  const int lane = tid & 31;
  const int w    = tid >> 5;
  const int hl   = lane >> 4;
  const int l16  = lane & 15;

  __shared__ __attribute__((aligned(16))) _Float16 Ksh[2][64 * KS];
  __shared__ __attribute__((aligned(16))) _Float16 Vsh[2][128 * VS];
  __shared__ __attribute__((aligned(16))) _Float16 Psh[8][16 * PS];

  // Q A-fragment for this wave's 16 rows (real K=16, upper K=16..31 zero-pad)
  const int nrow = n0 + w * 16 + l16;
  half16 qa = {};
  {
    half8 qlo = *(const half8*)&qh[((size_t)b * NPOS + nrow) * CQ + 8 * hl];
#pragma unroll
    for (int i = 0; i < 8; ++i) qa[i] = qlo[i];
  }

  f32x8 oacc[8];
  {
    f32x8 z = {};
#pragma unroll
    for (int cc = 0; cc < 8; ++cc) oacc[cc] = z;
  }

  // prologue: stage chunk 0 into buffer 0
  stage_chunk(kh, vh, b, 0, Ksh[0], Vsh[0], tid);
  wait_async();
  __syncthreads();

  int cur = 0;
  for (int mc = 0; mc < NPOS; mc += 64) {
    // kick off async staging of the next chunk into the other buffer
    if (mc + 64 < NPOS)
      stage_chunk(kh, vh, b, mc + 64, Ksh[cur ^ 1], Vsh[cur ^ 1], tid);

    const _Float16* Kc = Ksh[cur];
    const _Float16* Vc = Vsh[cur];

#pragma unroll
    for (int ms = 0; ms < 64; ms += 32) {
      // K B-fragments for two 16-col m tiles. Valid K=8*hl+0..7 lives in
      // elements 0-7; elements 8-15 (K=16..31) are the zero padding -> stay zero.
      half16 kb0 = {}, kb1 = {};
      {
        const half8* r0 = (const half8*)&Kc[(ms + l16) * KS + 8 * hl];
        const half8* r1 = (const half8*)&Kc[(ms + 16 + l16) * KS + 8 * hl];
#pragma unroll
        for (int i = 0; i < 8; ++i) {
          kb0[i] = r0[0][i];
          kb1[i] = r1[0][i];
        }
      }
      f32x8 z = {};
      f32x8 s0 = __builtin_amdgcn_wmma_f32_16x16x32_f16(false, qa, false, kb0,
                                                        (short)0, z, false, false);
      f32x8 s1 = __builtin_amdgcn_wmma_f32_16x16x32_f16(false, qa, false, kb1,
                                                        (short)0, z, false, false);
      // sigmoid gate, park P (16n x 32m) in per-wave LDS to re-layout as B
#pragma unroll
      for (int r = 0; r < 8; ++r) {
        const float p0 = __builtin_amdgcn_rcpf(1.0f + __expf(-s0[r]));
        const float p1 = __builtin_amdgcn_rcpf(1.0f + __expf(-s1[r]));
        Psh[w][(r + 8 * hl) * PS + l16]      = (_Float16)p0;
        Psh[w][(r + 8 * hl) * PS + 16 + l16] = (_Float16)p1;
      }
      // P^T B-fragment: lane = col n (l16), K = m = {8*hl..+7, 16+8*hl..+7}
      half8 plo = *(const half8*)&Psh[w][l16 * PS + 8 * hl];
      half8 phi = *(const half8*)&Psh[w][l16 * PS + 16 + 8 * hl];
      half16 pb = cat16(plo, phi);
      // O += V(16c x 32m) * P^T(32m x 16n) over all 8 channel tiles
#pragma unroll
      for (int cc = 0; cc < 8; ++cc) {
        const int c = cc * 16 + l16;
        half8 vlo = *(const half8*)&Vc[c * VS + ms + 8 * hl];
        half8 vhi = *(const half8*)&Vc[c * VS + ms + 16 + 8 * hl];
        oacc[cc] = __builtin_amdgcn_wmma_f32_16x16x32_f16(false, cat16(vlo, vhi),
                                                          false, pb, (short)0,
                                                          oacc[cc], false, false);
      }
    }

    // publish next buffer (async writes done) + retire current buffer reads
    wait_async();
    __syncthreads();
    cur ^= 1;
  }

  // epilogue: out = gamma * O + x
  const float g = gamma[0];
#pragma unroll
  for (int cc = 0; cc < 8; ++cc) {
#pragma unroll
    for (int r = 0; r < 8; ++r) {
      const int c = cc * 16 + r + 8 * hl;
      const size_t off = ((size_t)b * CCH + c) * NPOS + nrow;
      out[off] = g * oacc[cc][r] + x[off];
    }
  }
}

// ---------------------------------------------------------------------------
extern "C" void kernel_launch(void* const* d_in, const int* in_sizes, int n_in,
                              void* d_out, int out_size, void* d_ws, size_t ws_size,
                              hipStream_t stream) {
  const float* x     = (const float*)d_in[0];
  const float* Wq    = (const float*)d_in[1];
  const float* bq    = (const float*)d_in[2];
  const float* Wk    = (const float*)d_in[3];
  const float* bk    = (const float*)d_in[4];
  const float* Wv    = (const float*)d_in[5];
  const float* bv    = (const float*)d_in[6];
  const float* gamma = (const float*)d_in[7];
  float* out = (float*)d_out;

  // workspace: qh (512KB) | kh (512KB) | vh (4MB)  — all f16
  _Float16* qh = (_Float16*)d_ws;
  _Float16* kh = qh + (size_t)4 * NPOS * CQ;
  _Float16* vh = kh + (size_t)4 * NPOS * CQ;

  proj_kernel<<<4 * (NPOS / 64), 256, 0, stream>>>(x, Wq, bq, Wk, bk, Wv, bv,
                                                   qh, kh, vh);
  attn_kernel<<<4 * (NPOS / 128), 256, 0, stream>>>(x, qh, kh, vh, gamma, out);
}
